// ScaleDotProductAttention_33621003993261
// MI455X (gfx1250) — matrix operand order repaired and checked by reading, not compiled, not run on hardware
//
#include <hip/hip_runtime.h>

typedef _Float16 v16h __attribute__((ext_vector_type(16)));
typedef float    v8f  __attribute__((ext_vector_type(8)));

#define BB 4
#define HH 16
#define SS 1024
#define DD 128
#define QROWS 32                         // q-rows per workgroup (2 row-blocks of 16)
#define NEG_INF (-1.0e9f)
#define SCALE 0.08838834764831845f       // 1/sqrt(128)
#define LOG2E 1.4426950408889634f
#define LDS_STRIDE 1028                  // 1024 + 4 floats pad -> conflict-free column reads

__global__ __launch_bounds__(256)
void sdpa_fused_kernel(const float* __restrict__ q, const float* __restrict__ k,
                       const float* __restrict__ v, const int* __restrict__ mask,
                       float* __restrict__ ctx_out, float* __restrict__ attn_out)
{
    __shared__ float sc[QROWS * LDS_STRIDE];   // 32 x 1024 score/prob stripe (f32), 131.5 KB
    __shared__ float red[QROWS * 8];           // per-row partial reductions
    __shared__ float rowstat[QROWS];           // row max, then row 1/sum

    const int tid  = threadIdx.x;
    const int lane = tid & 31;
    const int wave = tid >> 5;              // 0..7
    const int grp  = lane >> 4;             // lane group 0/1
    const int ln   = lane & 15;             // 0..15

    const int qt = blockIdx.x;              // q-tile 0..31
    const int bh = blockIdx.y;              // 0..63
    const int b  = bh >> 4;                 // H = 16
    const int q0 = qt * QROWS;

    const float* qb = q + ((size_t)bh * SS + q0) * DD;
    const float* kb = k + (size_t)bh * SS * DD;
    const float* vb = v + (size_t)bh * SS * DD;

    // ================= Phase 1: scores = mask(scale * Q K^T) -> LDS =================
    // A fragments for both 16-row blocks (each 16x128 -> 4 chunks of 16x32 f16).
    // A layout used here (contiguous k): lane m=ln, element e: K = chunk*32 + grp*16 + e
    v16h A0[4], A1[4];
    {
        const float* qr0 = qb + ln * DD + grp * 16;            // rows 0..15
        const float* qr1 = qb + (16 + ln) * DD + grp * 16;     // rows 16..31
        #pragma unroll
        for (int c = 0; c < 4; ++c) {
            const float* p0 = qr0 + c * 32;
            const float* p1 = qr1 + c * 32;
            #pragma unroll
            for (int e = 0; e < 8; ++e) {
                A0[c][e]     = (_Float16)p0[e];
                A0[c][e + 8] = (_Float16)p0[e + 8];
                A1[c][e]     = (_Float16)p1[e];
                A1[c][e + 8] = (_Float16)p1[e + 8];
            }
        }
    }

    // Each wave: 128 columns = 8 n-tiles of 16; each K fragment feeds BOTH row blocks.
    #pragma unroll 1
    for (int t = 0; t < 8; ++t) {
        const int n0 = wave * 128 + t * 16;
        v8f acc0 = {0.f, 0.f, 0.f, 0.f, 0.f, 0.f, 0.f, 0.f};
        v8f acc1 = {0.f, 0.f, 0.f, 0.f, 0.f, 0.f, 0.f, 0.f};
        // B fragment from K rows (K^T): lane column n = n0+ln is row n of K.
        // B layout used here (same contiguous k as A): element e: Kdim = chunk*32 + grp*16 + e (16 consecutive floats)
        const float* kr = kb + (size_t)(n0 + ln) * DD + grp * 16;
        #pragma unroll
        for (int c = 0; c < 4; ++c) {
            v16h bf;
            const float* p = kr + c * 32;
            #pragma unroll
            for (int e = 0; e < 16; ++e) bf[e] = (_Float16)p[e];
            acc0 = __builtin_amdgcn_wmma_f32_16x16x32_f16(
                       false, A0[c], false, bf, (short)0, acc0, false, false);
            acc1 = __builtin_amdgcn_wmma_f32_16x16x32_f16(
                       false, A1[c], false, bf, (short)0, acc1, false, false);
        }
        // C layout: VGPR r -> row M = r + 8*grp, col N = ln. Apply scale + mask.
        #pragma unroll
        for (int r = 0; r < 8; ++r) {
            const int mo = r + 8 * grp;
            const int mv0 = mask[((size_t)b * SS + q0 + mo) * SS + n0 + ln];
            const int mv1 = mask[((size_t)b * SS + q0 + 16 + mo) * SS + n0 + ln];
            sc[mo * LDS_STRIDE + n0 + ln]        = (mv0 == 0) ? NEG_INF : acc0[r] * SCALE;
            sc[(16 + mo) * LDS_STRIDE + n0 + ln] = (mv1 == 0) ? NEG_INF : acc1[r] * SCALE;
        }
    }
    __syncthreads();

    // ================= Phase 2: softmax over each row of 1024, write attn =================
    {
        const int row = tid >> 3;           // 0..31
        const int seg = tid & 7;            // 0..7, 128 cols each
        float* sr = &sc[row * LDS_STRIDE + seg * 128];

        float mx = -3.4e38f;
        #pragma unroll 4
        for (int j = 0; j < 128; ++j) mx = fmaxf(mx, sr[j]);
        red[row * 8 + seg] = mx;
        __syncthreads();
        if (seg == 0) {
            float m = red[row * 8];
            #pragma unroll
            for (int j = 1; j < 8; ++j) m = fmaxf(m, red[row * 8 + j]);
            rowstat[row] = m;
        }
        __syncthreads();
        const float rm = rowstat[row];
        float sum = 0.0f;
        #pragma unroll 4
        for (int j = 0; j < 128; ++j) {
            const float e = __builtin_exp2f((sr[j] - rm) * LOG2E);
            sr[j] = e;
            sum += e;
        }
        red[row * 8 + seg] = sum;
        __syncthreads();
        if (seg == 0) {
            float s = red[row * 8];
            #pragma unroll
            for (int j = 1; j < 8; ++j) s += red[row * 8 + j];
            rowstat[row] = 1.0f / s;
        }
        __syncthreads();
        // normalize in LDS + coalesced streaming write of attn tile (written exactly once)
        float* attn_b = attn_out + ((size_t)bh * SS + q0) * SS;
        for (int idx = tid; idx < QROWS * 1024; idx += 256) {
            const int r = idx >> 10, c = idx & 1023;
            const float pv = sc[r * LDS_STRIDE + c] * rowstat[r];
            sc[r * LDS_STRIDE + c] = pv;
            attn_b[(size_t)r * SS + c] = pv;
        }
        __syncthreads();
    }

    // ================= Phase 3: context = P @ V (32x1024 @ 1024x128) =================
    {
        v8f acc0 = {0.f, 0.f, 0.f, 0.f, 0.f, 0.f, 0.f, 0.f};   // rows 0..15
        v8f acc1 = {0.f, 0.f, 0.f, 0.f, 0.f, 0.f, 0.f, 0.f};   // rows 16..31
        const int col0 = wave * 16;         // this wave owns output cols [col0, col0+16)

        #pragma unroll 1
        for (int kc = 0; kc < 32; ++kc) {
            const int k0 = kc * 32;
            // A fragments = P chunks from LDS for both row blocks
            v16h a0, a1;
            const float* pp0 = &sc[ln * LDS_STRIDE + k0 + grp * 16];
            const float* pp1 = &sc[(16 + ln) * LDS_STRIDE + k0 + grp * 16];
            #pragma unroll
            for (int e = 0; e < 8; ++e) {
                a0[e]     = (_Float16)pp0[e];
                a0[e + 8] = (_Float16)pp0[e + 8];
                a1[e]     = (_Float16)pp1[e];
                a1[e + 8] = (_Float16)pp1[e + 8];
            }
            // One B fragment from V feeds BOTH row blocks:
            // elem e -> V[k0 + grp*16 + e][col0+ln]; 64B coalesced per lane group
            const float* vr = vb + (size_t)(k0 + grp * 16) * DD + col0 + ln;
            v16h bf;
            #pragma unroll
            for (int e = 0; e < 16; ++e) bf[e] = (_Float16)vr[(size_t)e * DD];
            acc0 = __builtin_amdgcn_wmma_f32_16x16x32_f16(
                       false, a0, false, bf, (short)0, acc0, false, false);
            acc1 = __builtin_amdgcn_wmma_f32_16x16x32_f16(
                       false, a1, false, bf, (short)0, acc1, false, false);
        }
        float* cb = ctx_out + ((size_t)bh * SS + q0) * DD;
        #pragma unroll
        for (int r = 0; r < 8; ++r) {
            const int mo = r + 8 * grp;
            cb[(size_t)mo * DD + col0 + ln]        = acc0[r];
            cb[(size_t)(16 + mo) * DD + col0 + ln] = acc1[r];
        }
    }
}

extern "C" void kernel_launch(void* const* d_in, const int* in_sizes, int n_in,
                              void* d_out, int out_size, void* d_ws, size_t ws_size,
                              hipStream_t stream) {
    const float* q    = (const float*)d_in[0];
    const float* k    = (const float*)d_in[1];
    const float* v    = (const float*)d_in[2];
    const int*   mask = (const int*)d_in[3];

    float* ctx  = (float*)d_out;                                   // [B,H,S,D]
    float* attn = (float*)d_out + (size_t)BB * HH * SS * DD;       // [B,H,S,S]

    dim3 grid(SS / QROWS, BB * HH);   // 32 q-tiles x 64 (b,h)
    dim3 block(256);                  // 8 waves
    sdpa_fused_kernel<<<grid, block, 0, stream>>>(q, k, v, mask, ctx, attn);
}
